// SparseAttention3d_41128606826831
// MI455X (gfx1250) — matrix operand order repaired and checked by reading, not compiled, not run on hardware
//
#include <hip/hip_runtime.h>
#include <hip/hip_bf16.h>

typedef __attribute__((ext_vector_type(16))) __bf16 v16bf;
typedef __attribute__((ext_vector_type(8)))  float  v8f;

#define NVX 40000
#define NQX 8192
#define KKX 96
#define CX  128
#define FFX 256
#define HX  8
#define DHX 16

__device__ __forceinline__ unsigned short f2bf(float f) {
  unsigned int u = __float_as_uint(f);
  u += 0x7FFFu + ((u >> 16) & 1u);          // round-to-nearest-even
  return (unsigned short)(u >> 16);
}
__device__ __forceinline__ float bf2f(unsigned short h) {
  return __uint_as_float(((unsigned int)h) << 16);
}

// -------- rowwise LayerNorm (C=128), one wave per row, bf16 output --------
__global__ __launch_bounds__(256)
void ln_rows_kernel(const float* __restrict__ x, const float* __restrict__ g,
                    const float* __restrict__ b, unsigned short* __restrict__ out,
                    int nrows) {
  int wave = threadIdx.x >> 5;
  int lane = threadIdx.x & 31;
  int row  = blockIdx.x * 8 + wave;
  if (row >= nrows) return;
  const float* xr = x + (size_t)row * CX;
  float v0 = xr[lane], v1 = xr[lane + 32], v2 = xr[lane + 64], v3 = xr[lane + 96];
  float s  = v0 + v1 + v2 + v3;
  float ss = v0*v0 + v1*v1 + v2*v2 + v3*v3;
  for (int off = 16; off; off >>= 1) { s += __shfl_xor(s, off); ss += __shfl_xor(ss, off); }
  float mean = s * (1.0f / CX);
  float var  = ss * (1.0f / CX) - mean * mean;
  float inv  = rsqrtf(var + 1e-5f);
  unsigned short* orow = out + (size_t)row * CX;
  orow[lane]      = f2bf((v0 - mean) * inv * g[lane]      + b[lane]);
  orow[lane + 32] = f2bf((v1 - mean) * inv * g[lane + 32] + b[lane + 32]);
  orow[lane + 64] = f2bf((v2 - mean) * inv * g[lane + 64] + b[lane + 64]);
  orow[lane + 96] = f2bf((v3 - mean) * inv * g[lane + 96] + b[lane + 96]);
}

// -------- swizzle fp32 weight W[N][Kdim] into WMMA B-fragment bf16 layout ---
// fragment f = s*(N/16)+nt holds B[K=s*32..+32)[N=nt*16..+16); per lane 32B:
// 16 contiguous bf16 = K (lane>>4)*16+j for column n = lane&15.
__global__ void swz_kernel(const float* __restrict__ W, unsigned short* __restrict__ out,
                           int N, int Kdim) {
  int u = blockIdx.x * blockDim.x + threadIdx.x;
  if (u >= N * Kdim) return;
  int j = u & 15, lane = (u >> 4) & 31, f = u >> 9;
  int ntiles = N >> 4;
  int nt = f % ntiles, s = f / ntiles;
  int n = nt * 16 + (lane & 15);
  int k = s * 32 + ((lane >> 4) << 4) + j;
  out[u] = f2bf(W[(size_t)n * Kdim + k]);
}

// -------- q_feat = relu(query_coords @ q_pos_w.T + b), bf16 out ------------
__global__ void qfeat_kernel(const float* __restrict__ qc, const float* __restrict__ w,
                             const float* __restrict__ b, unsigned short* __restrict__ out) {
  int e = blockIdx.x * 256 + threadIdx.x;
  if (e >= NQX * CX) return;
  int n = e >> 7, c = e & 127;
  const float* q  = qc + (size_t)n * 3;
  const float* wr = w + (size_t)c * 3;
  float v = fmaf(q[0], wr[0], fmaf(q[1], wr[1], fmaf(q[2], wr[2], b[c])));
  out[e] = f2bf(fmaxf(v, 0.0f));
}

// -------- generic 32-row-tile bf16 WMMA GEMM: out = act(A@W.T + bias [+res])
// A staged into LDS via GLOBAL_LOAD_ASYNC_TO_LDS_B128 (ASYNCcnt path).
// Each wave: one (or strided) N-tile, TWO M-tiles sharing each B fragment ->
// two independent WMMA accumulation chains, no dependent-WMMA stalls.
__global__ __launch_bounds__(256)
void gemm_bf16_kernel(const unsigned short* __restrict__ A,
                      const unsigned short* __restrict__ Bswz,
                      const float* __restrict__ bias, const float* __restrict__ resid,
                      float* __restrict__ outF, unsigned short* __restrict__ outBf,
                      int N, int Ka, int doRelu) {
  extern __shared__ unsigned short As[];   // 32 rows x Ka bf16, dynamic => LDS offset 0
  int tid = threadIdx.x;
  int mblk = blockIdx.x;                   // 32-row block
  {
    const unsigned short* src = A + (size_t)mblk * 32 * Ka;
    int chunks = (32 * Ka) >> 3;           // 16B chunks
    for (int i = tid; i < chunks; i += 256) {
      unsigned lds = (unsigned)(i << 4);   // raw LDS byte offset (dynamic base = 0)
      unsigned long long ga = (unsigned long long)(src + (i << 3));
      asm volatile("global_load_async_to_lds_b128 %0, %1, off"
                   :: "v"(lds), "v"(ga) : "memory");
    }
    asm volatile("s_wait_asynccnt 0" ::: "memory");
  }
  __syncthreads();

  int wave = tid >> 5, lane = tid & 31;
  int m = lane & 15, half = lane >> 4;
  int ntiles = N >> 4, ksteps = Ka >> 5;
  for (int nt = wave; nt < ntiles; nt += 8) {
    float bv = bias[nt * 16 + m];
    v8f acc0 = {bv, bv, bv, bv, bv, bv, bv, bv};
    v8f acc1 = acc0;
    for (int s = 0; s < ksteps; ++s) {
      v16bf bfr, a0, a1;
      const uint4* bp = (const uint4*)(Bswz + (((size_t)(s * ntiles + nt) * 32 + lane) << 4));
      ((uint4*)&bfr)[0] = bp[0];
      ((uint4*)&bfr)[1] = bp[1];
      const unsigned short* ap0 = As + m * Ka + s * 32 + half * 16;
      const unsigned short* ap1 = ap0 + 16 * Ka;
      ((uint4*)&a0)[0] = *(const uint4*)(ap0);
      ((uint4*)&a0)[1] = *(const uint4*)(ap0 + 8);
      ((uint4*)&a1)[0] = *(const uint4*)(ap1);
      ((uint4*)&a1)[1] = *(const uint4*)(ap1 + 8);
      acc0 = __builtin_amdgcn_wmma_f32_16x16x32_bf16(false, a0, false, bfr,
                                                     (short)0, acc0, false, false);
      acc1 = __builtin_amdgcn_wmma_f32_16x16x32_bf16(false, a1, false, bfr,
                                                     (short)0, acc1, false, false);
    }
    int col = nt * 16 + m;
#pragma unroll
    for (int t = 0; t < 2; ++t) {
      v8f acc = t ? acc1 : acc0;
#pragma unroll
      for (int r = 0; r < 8; ++r) {
        int row = mblk * 32 + t * 16 + half * 8 + r;
        size_t o = (size_t)row * N + col;
        float v = acc[r];
        if (resid) v += resid[o];
        if (doRelu) v = fmaxf(v, 0.0f);
        if (outF)  outF[o]  = v;
        if (outBf) outBf[o] = f2bf(v);
      }
    }
  }
}

// -------- fused sparse attention: 1 query per block, 256 threads -----------
// LDS: kf(96x128 bf16) | K(96x128 f32) | V(96x128 f32) | scores | q | kposw...
#define SM_KF   0
#define SM_K    24576
#define SM_V    73728
#define SM_AT   122880
#define SM_Q    125952
#define SM_KW   126464
#define SM_KB   128000
#define SM_CD   128512
#define SM_IDX  129664
#define SM_SIZE 130048

__global__ __launch_bounds__(256)
void attn_kernel(const unsigned short* __restrict__ vf_bf,
                 const float* __restrict__ vcoords,
                 const int* __restrict__ kidx,
                 const float* __restrict__ kposw, const float* __restrict__ kposb,
                 const unsigned short* __restrict__ wk_swz,
                 const unsigned short* __restrict__ wv_swz,
                 const float* __restrict__ in_b,
                 const float* __restrict__ qvec,
                 unsigned short* __restrict__ ctx_bf) {
  extern __shared__ unsigned char smem[];
  unsigned short* s_kf = (unsigned short*)(smem + SM_KF);
  float* s_k  = (float*)(smem + SM_K);
  float* s_v  = (float*)(smem + SM_V);
  float* s_at = (float*)(smem + SM_AT);
  float* s_q  = (float*)(smem + SM_Q);
  float* s_kw = (float*)(smem + SM_KW);
  float* s_kb = (float*)(smem + SM_KB);
  float* s_cd = (float*)(smem + SM_CD);
  int*   s_idx = (int*)(smem + SM_IDX);

  int n = blockIdx.x;
  int tid = threadIdx.x;

  if (tid < KKX)     s_idx[tid] = kidx[(size_t)n * KKX + tid];
  if (tid < CX)      { s_q[tid] = qvec[(size_t)n * CX + tid]; s_kb[tid] = kposb[tid]; }
  if (tid < 3 * CX)  s_kw[tid] = kposw[tid];
  __syncthreads();
  if (tid < 3 * KKX) {
    int kk = tid / 3, t = tid - kk * 3;
    int g = s_idx[kk]; if (g < 0) g = 0;
    s_cd[tid] = vcoords[(size_t)g * 3 + t];
  }
  __syncthreads();

  // gather normed voxel features + relu positional projection, store bf16
  for (int e = tid; e < KKX * CX; e += 256) {
    int kk = e >> 7, c = e & 127;
    int g = s_idx[kk]; if (g < 0) g = 0;
    float feat = bf2f(vf_bf[(size_t)g * CX + c]);
    const float* cd = s_cd + kk * 3;
    const float* w  = s_kw + c * 3;
    float pos = fmaf(cd[0], w[0], fmaf(cd[1], w[1], fmaf(cd[2], w[2], s_kb[c])));
    s_kf[e] = f2bf(feat + fmaxf(pos, 0.0f));
  }
  __syncthreads();

  // k AND v projections fused: wave owns N-tile nt=wave; per k-step load one
  // B_k frag + one B_v frag, then for each of 6 M-tiles load A once and issue
  // 2 independent WMMAs (12 independent accumulation chains per wave).
  int wave = tid >> 5, lane = tid & 31;
  int ml = lane & 15, half = lane >> 4;
  {
    int nt = wave;                          // 8 waves == 8 n-tiles
    float bk0 = in_b[CX + nt * 16 + ml];
    float bv0 = in_b[2 * CX + nt * 16 + ml];
    v8f ak[6], av[6];
#pragma unroll
    for (int mt = 0; mt < 6; ++mt) {
      v8f ik = {bk0, bk0, bk0, bk0, bk0, bk0, bk0, bk0};
      v8f iv = {bv0, bv0, bv0, bv0, bv0, bv0, bv0, bv0};
      ak[mt] = ik; av[mt] = iv;
    }
#pragma unroll
    for (int s = 0; s < 4; ++s) {
      v16bf bk, bv;
      size_t foff = ((size_t)((s * 8 + nt) * 32 + lane) << 4);
      const uint4* bpk = (const uint4*)(wk_swz + foff);
      const uint4* bpv = (const uint4*)(wv_swz + foff);
      ((uint4*)&bk)[0] = bpk[0];
      ((uint4*)&bk)[1] = bpk[1];
      ((uint4*)&bv)[0] = bpv[0];
      ((uint4*)&bv)[1] = bpv[1];
#pragma unroll
      for (int mt = 0; mt < 6; ++mt) {
        v16bf a;
        const unsigned short* ap = s_kf + (mt * 16 + ml) * CX + s * 32 + half * 16;
        ((uint4*)&a)[0] = *(const uint4*)(ap);
        ((uint4*)&a)[1] = *(const uint4*)(ap + 8);
        ak[mt] = __builtin_amdgcn_wmma_f32_16x16x32_bf16(false, a, false, bk,
                                                         (short)0, ak[mt], false, false);
        av[mt] = __builtin_amdgcn_wmma_f32_16x16x32_bf16(false, a, false, bv,
                                                         (short)0, av[mt], false, false);
      }
    }
    int col = nt * 16 + ml;
#pragma unroll
    for (int mt = 0; mt < 6; ++mt) {
#pragma unroll
      for (int r = 0; r < 8; ++r) {
        int row = mt * 16 + half * 8 + r;
        s_k[row * CX + col] = ak[mt][r];
        s_v[row * CX + col] = av[mt][r];
      }
    }
  }
  __syncthreads();

  // scores (8 heads x 96 keys), scale 1/sqrt(16)=0.25, mask invalid keys
  for (int e = tid; e < HX * KKX; e += 256) {
    int h = e / KKX, kk = e - h * KKX;
    float sc;
    if (s_idx[kk] < 0) sc = -1e9f;
    else {
      const float* qh = s_q + h * DHX;
      const float* kr = s_k + kk * CX + h * DHX;
      float d = 0.0f;
#pragma unroll
      for (int j = 0; j < DHX; ++j) d = fmaf(qh[j], kr[j], d);
      sc = d * 0.25f;
    }
    s_at[e] = sc;
  }
  __syncthreads();

  // softmax over 96 keys: wave h owns head h (3 keys / lane, shuffle reduce)
  {
    int h = wave;
    float a0 = s_at[h * KKX + lane];
    float a1 = s_at[h * KKX + lane + 32];
    float a2 = s_at[h * KKX + lane + 64];
    float mx = fmaxf(a0, fmaxf(a1, a2));
    for (int off = 16; off; off >>= 1) mx = fmaxf(mx, __shfl_xor(mx, off));
    float e0 = __expf(a0 - mx), e1 = __expf(a1 - mx), e2 = __expf(a2 - mx);
    float s = e0 + e1 + e2;
    for (int off = 16; off; off >>= 1) s += __shfl_xor(s, off);
    float inv = 1.0f / s;
    s_at[h * KKX + lane]      = e0 * inv;
    s_at[h * KKX + lane + 32] = e1 * inv;
    s_at[h * KKX + lane + 64] = e2 * inv;
  }
  __syncthreads();

  // ctx = attn @ V  -> bf16 (feeds WMMA out-projection GEMM)
  if (tid < CX) {
    int h = tid >> 4, d = tid & 15;
    const float* at = s_at + h * KKX;
    const float* vc = s_v + h * DHX + d;
    float acc = 0.0f;
#pragma unroll 4
    for (int kk = 0; kk < KKX; ++kk) acc = fmaf(at[kk], vc[kk * CX], acc);
    ctx_bf[(size_t)n * CX + tid] = f2bf(acc);
  }
}

extern "C" void kernel_launch(void* const* d_in, const int* in_sizes, int n_in,
                              void* d_out, int out_size, void* d_ws, size_t ws_size,
                              hipStream_t stream) {
  const float* voxel_features = (const float*)d_in[0];
  const float* voxel_coords   = (const float*)d_in[1];
  const float* query_coords   = (const float*)d_in[2];
  const int*   key_indices    = (const int*)d_in[3];
  const float* norm1_g = (const float*)d_in[4];
  const float* norm1_b = (const float*)d_in[5];
  const float* q_pos_w = (const float*)d_in[6];
  const float* q_pos_b = (const float*)d_in[7];
  const float* k_pos_w = (const float*)d_in[8];
  const float* k_pos_b = (const float*)d_in[9];
  const float* in_w    = (const float*)d_in[10];
  const float* in_b    = (const float*)d_in[11];
  const float* out_w   = (const float*)d_in[12];
  const float* out_b   = (const float*)d_in[13];
  const float* norm2_g = (const float*)d_in[14];
  const float* norm2_b = (const float*)d_in[15];
  const float* lin1_w  = (const float*)d_in[16];
  const float* lin1_b  = (const float*)d_in[17];
  const float* lin2_w  = (const float*)d_in[18];
  const float* lin2_b  = (const float*)d_in[19];
  const float* fin_w   = (const float*)d_in[20];
  const float* fin_b   = (const float*)d_in[21];
  float* out = (float*)d_out;
  (void)in_sizes; (void)n_in; (void)out_size; (void)ws_size;

  unsigned char* ws = (unsigned char*)d_ws;
  size_t off = 0;
  auto alloc = [&](size_t bytes) {
    unsigned char* p = ws + off;
    off = (off + bytes + 255) & ~(size_t)255;
    return p;
  };
  unsigned short* vf_bf = (unsigned short*)alloc((size_t)NVX * CX * 2);
  unsigned short* qfeat = (unsigned short*)alloc((size_t)NQX * CX * 2);
  float*          qv    = (float*)         alloc((size_t)NQX * CX * 4);
  unsigned short* ctx   = (unsigned short*)alloc((size_t)NQX * CX * 2);
  float*          att   = (float*)         alloc((size_t)NQX * CX * 4);
  unsigned short* hbf   = (unsigned short*)alloc((size_t)NQX * CX * 2);
  unsigned short* ffb   = (unsigned short*)alloc((size_t)NQX * FFX * 2);
  unsigned short* xbf   = (unsigned short*)alloc((size_t)NQX * CX * 2);
  unsigned short* wq_s  = (unsigned short*)alloc((size_t)CX * CX * 2);
  unsigned short* wk_s  = (unsigned short*)alloc((size_t)CX * CX * 2);
  unsigned short* wv_s  = (unsigned short*)alloc((size_t)CX * CX * 2);
  unsigned short* wo_s  = (unsigned short*)alloc((size_t)CX * CX * 2);
  unsigned short* l1_s  = (unsigned short*)alloc((size_t)FFX * CX * 2);
  unsigned short* l2_s  = (unsigned short*)alloc((size_t)CX * FFX * 2);
  unsigned short* fn_s  = (unsigned short*)alloc((size_t)CX * CX * 2);

  // weight swizzles into WMMA B-fragment layout
  int g1 = (CX * CX + 255) / 256;
  int g2 = (FFX * CX + 255) / 256;
  swz_kernel<<<g1, 256, 0, stream>>>(in_w,               wq_s, CX, CX);
  swz_kernel<<<g1, 256, 0, stream>>>(in_w + CX * CX,     wk_s, CX, CX);
  swz_kernel<<<g1, 256, 0, stream>>>(in_w + 2 * CX * CX, wv_s, CX, CX);
  swz_kernel<<<g1, 256, 0, stream>>>(out_w,              wo_s, CX, CX);
  swz_kernel<<<g2, 256, 0, stream>>>(lin1_w,             l1_s, FFX, CX);
  swz_kernel<<<g2, 256, 0, stream>>>(lin2_w,             l2_s, CX, FFX);
  swz_kernel<<<g1, 256, 0, stream>>>(fin_w,              fn_s, CX, CX);

  // norm1(voxel_features) -> bf16 gather table
  ln_rows_kernel<<<NVX / 8, 256, 0, stream>>>(voxel_features, norm1_g, norm1_b, vf_bf, NVX);
  // q_feat = relu(coords @ q_pos_w.T + b)
  qfeat_kernel<<<(NQX * CX) / 256, 256, 0, stream>>>(query_coords, q_pos_w, q_pos_b, qfeat);
  // q = q_feat @ wq.T + bq   (f32)
  gemm_bf16_kernel<<<NQX / 32, 256, 32 * CX * 2, stream>>>(
      qfeat, wq_s, in_b, nullptr, qv, nullptr, CX, CX, 0);
  // fused gather + k/v projection (WMMA) + softmax + ctx
  attn_kernel<<<NQX, 256, SM_SIZE, stream>>>(vf_bf, voxel_coords, key_indices,
                                             k_pos_w, k_pos_b, wk_s, wv_s, in_b, qv, ctx);
  // attend = ctx @ out_w.T + out_b   (f32, kept for residual + LN)
  gemm_bf16_kernel<<<NQX / 32, 256, 32 * CX * 2, stream>>>(
      ctx, wo_s, out_b, nullptr, att, nullptr, CX, CX, 0);
  // h = norm2(attend) -> bf16
  ln_rows_kernel<<<NQX / 8, 256, 0, stream>>>(att, norm2_g, norm2_b, hbf, NQX);
  // ff = relu(h @ lin1.T + b1) -> bf16
  gemm_bf16_kernel<<<NQX / 32, 256, 32 * CX * 2, stream>>>(
      hbf, l1_s, lin1_b, nullptr, nullptr, ffb, FFX, CX, 1);
  // x = attend + (ff @ lin2.T + b2) -> bf16
  gemm_bf16_kernel<<<NQX / 32, 256, 32 * FFX * 2, stream>>>(
      ffb, l2_s, lin2_b, att, nullptr, xbf, CX, FFX, 0);
  // out = relu(x @ fin.T + fb) -> f32
  gemm_bf16_kernel<<<NQX / 32, 256, 32 * CX * 2, stream>>>(
      xbf, fn_s, fin_b, nullptr, out, nullptr, CX, CX, 1);
}
